// LSTMModel_32487132627077
// MI455X (gfx1250) — matrix operand order repaired and checked by reading, not compiled, not run on hardware
//
#include <hip/hip_runtime.h>

typedef _Float16 v16h __attribute__((ext_vector_type(16)));
typedef _Float16 v8h  __attribute__((ext_vector_type(8)));
typedef _Float16 v4h  __attribute__((ext_vector_type(4)));
typedef float    v8f  __attribute__((ext_vector_type(8)));

#define B_TOT 4096
#define T_LEN 256
#define D_IN  4

// B-matrix fragments pre-swizzled to WMMA lane layout.
// fragment = 32 lanes x 16 halves = 512 f16 = 1KB
#define FRAG_HALVES 512
#define W0_FRAGS   (16*3)           // 16 n-tiles x 3 k-chunks (K=96: [Whh0 | Wih0 | 1-col(bias) | 0])
#define W1_FRAGS   (16*4)           // 16 n-tiles x 4 k-chunks (K=128: [Wih1 | Whh1])
#define W0_HALVES  (W0_FRAGS*FRAG_HALVES)   // 24576
#define W1_HALVES  (W1_FRAGS*FRAG_HALVES)   // 32768
#define XB_ROW 32                   // layer0 A chunk2 source: [x(4) | 1.0 | zeros]
#define A1_ROW 128                  // [h0_t(64) | h1_prev(64)]; layer0 reads h0 from cols 0..63
#define XB_HALVES (16*XB_ROW)
#define A1_HALVES (16*A1_ROW)
#define SMEM_HALVES (W0_HALVES + W1_HALVES + XB_HALVES + A1_HALVES)
#define SMEM_BYTES  (SMEM_HALVES*2)   // 119808 B, fits 320KB WGP LDS

// Branchless activations on native trans ops (v_exp_f32 / v_rcp_f32):
// no EXEC divergence, co-executes with 16-bit (TRANS-class) WMMA.
__device__ __forceinline__ float fast_sigmoid(float x) {
  return __builtin_amdgcn_rcpf(1.0f + __expf(-x));
}
__device__ __forceinline__ float fast_tanh(float x) {
  // tanh(x) = 1 - 2/(exp(2x)+1); rcp(inf)=0 -> +1, rcp(1) -> -1 (correct saturation)
  return 1.0f - 2.0f * __builtin_amdgcn_rcpf(__expf(2.0f * x) + 1.0f);
}

// Compiler-only ordering barrier: same-wave LDS ops execute in order at the HW
// (ISA 7.3), so store->load visibility needs no DScnt drain, just program order.
__device__ __forceinline__ void lds_order() {
  asm volatile("" ::: "memory");
}

// Build a v16h fragment from two 16-byte LDS reads (ds_load_b128 x2).
__device__ __forceinline__ v16h ld_frag(const _Float16* p0, const _Float16* p1) {
  v8h lo = *(const v8h*)p0;
  v8h hi = *(const v8h*)p1;
  v16h r;
#pragma unroll
  for (int i = 0; i < 8; ++i) { r[i] = lo[i]; r[i + 8] = hi[i]; }
  return r;
}
__device__ __forceinline__ v16h ld_frag2(const _Float16* p) { return ld_frag(p, p + 8); }

__global__ void __launch_bounds__(32)
lstm_fused_wmma(const float* __restrict__ x,
                const float* __restrict__ Wih0, const float* __restrict__ Whh0,
                const float* __restrict__ bih0, const float* __restrict__ bhh0,
                const float* __restrict__ Wih1, const float* __restrict__ Whh1,
                const float* __restrict__ bih1, const float* __restrict__ bhh1,
                const float* __restrict__ Wfc,  const float* __restrict__ bfc,
                float* __restrict__ out)
{
  extern __shared__ _Float16 smem[];
  _Float16* w0f  = smem;                    // layer0 B fragments
  _Float16* w1f  = w0f + W0_HALVES;         // layer1 B fragments
  _Float16* xbuf = w1f + W1_HALVES;         // [16][32]: x_t + ones column
  _Float16* a1   = xbuf + XB_HALVES;        // [16][128]: h0_t | h1_prev

  const int lane = threadIdx.x & 31;
  const int nl   = lane & 15;
  const int hi8  = (lane & 16) ? 8 : 0;

  // ---- one-time: swizzle weights into B-fragment layout ----
  // fragment (j=n-tile, c=k-chunk): lane l holds column n = j*16 + (l&15);
  // halves e=0..15 map to k = c*32 + ((l&16)?16:0) + e.   (contiguous k order, A uses the same)
  for (int idx = lane; idx < W0_HALVES; idx += 32) {
    int e = idx & 15;
    int l = (idx >> 4) & 31;
    int f = idx >> 9;
    int j = f / 3, c = f - j * 3;
    int n = j * 16 + (l & 15);
    int k = c * 32 + ((l & 16) ? 16 : 0) + e;
    float v = 0.0f;
    if (k < 64)       v = Whh0[n * 64 + k];
    else if (k < 68)  v = Wih0[n * 4 + (k - 64)];
    else if (k == 68) v = bih0[n] + bhh0[n];   // layer0 bias rides the ones-column
    w0f[idx] = (_Float16)v;
  }
  for (int idx = lane; idx < W1_HALVES; idx += 32) {
    int e = idx & 15;
    int l = (idx >> 4) & 31;
    int f = idx >> 9;
    int j = f >> 2, c = f & 3;
    int n = j * 16 + (l & 15);
    int k = c * 32 + ((l & 16) ? 16 : 0) + e;
    float v = (k < 64) ? Wih1[n * 64 + k] : Whh1[n * 64 + (k - 64)];
    w1f[idx] = (_Float16)v;
  }
  // zero state buffers; set ones-column (k=68 -> xbuf col 4)
  for (int idx = lane; idx < XB_HALVES + A1_HALVES; idx += 32)
    xbuf[idx] = (_Float16)0.0f;
  lds_order();
  if (lane < 16) xbuf[lane * XB_ROW + 4] = (_Float16)1.0f;

  // ---- layer1 bias as SCALAR regs (16 VGPRs, added in elementwise) ----
  float bias1[16];
#pragma unroll
  for (int j = 0; j < 16; ++j) bias1[j] = bih1[j * 16 + nl] + bhh1[j * 16 + nl];

  // cell state in registers: slot jj*8+r <-> (M = r + hi8, h_idx = jj*16 + nl)
  float c0[32], c1[32];
#pragma unroll
  for (int i = 0; i < 32; ++i) { c0[i] = 0.0f; c1[i] = 0.0f; }

  const int rowbase = blockIdx.x * 16;
  const v8f zero = {};

  for (int t = 0; t < T_LEN; ++t) {
    // stage x_t (4 floats/row) into xbuf cols 0..3 as f16
    if (lane < 16) {
      const float4 xv = *(const float4*)(x + ((size_t)(rowbase + lane) * T_LEN + t) * D_IN);
      v4h xh;
      xh[0] = (_Float16)xv.x; xh[1] = (_Float16)xv.y;
      xh[2] = (_Float16)xv.z; xh[3] = (_Float16)xv.w;
      *(v4h*)(xbuf + lane * XB_ROW) = xh;
      if (t + 1 < T_LEN)
        __builtin_prefetch(x + ((size_t)(rowbase + lane) * T_LEN + (t + 1)) * D_IN, 0, 1);
    }
    lds_order();

    // ---- layer 0 A fragments: chunks 0,1 = h0_prev (a1 cols 0..63), chunk 2 = [x|1|0] ----
    v16h A0[3];
#pragma unroll
    for (int c = 0; c < 2; ++c)
      A0[c] = ld_frag2(a1 + nl * A1_ROW + c * 32 + 2 * hi8);
    A0[2] = ld_frag2(xbuf + nl * XB_ROW + 2 * hi8);

#pragma unroll
    for (int jj = 0; jj < 4; ++jj) {   // h-tile jj -> gates i,f,g,o at n-tiles jj, jj+4, jj+8, jj+12
      v8f acc[4];
      // 12 WMMAs with a depth-2 B-fragment software pipeline (3-buffer ring):
      // fragment for step i+2 is loaded before WMMA i issues -> ds latency hidden.
      {
        v16h bA = ld_frag2(w0f + (size_t)((jj * 3 + 0) * 32 + lane) * 16);
        v16h bB = ld_frag2(w0f + (size_t)((jj * 3 + 1) * 32 + lane) * 16);
#pragma unroll
        for (int i = 0; i < 12; ++i) {
          const int g = i / 3, c = i - g * 3;
          v16h bN = bA;  // dead placeholder past the end
          if (i + 2 < 12) {
            const int i2 = i + 2, g2 = i2 / 3, c2 = i2 - g2 * 3;
            bN = ld_frag2(w0f + (size_t)(((jj + g2 * 4) * 3 + c2) * 32 + lane) * 16);
          }
          acc[g] = __builtin_amdgcn_wmma_f32_16x16x32_f16(
              false, A0[c], false, bA, (short)0, (c == 0) ? zero : acc[g], false, false);
          bA = bB; bB = bN;
        }
      }
#pragma unroll
      for (int r = 0; r < 8; ++r) {
        float iv = fast_sigmoid(acc[0][r]);
        float fv = fast_sigmoid(acc[1][r]);
        float gv = fast_tanh(acc[2][r]);
        float ov = fast_sigmoid(acc[3][r]);
        float cc = fv * c0[jj * 8 + r] + iv * gv;
        c0[jj * 8 + r] = cc;
        // h0_t -> a1 cols 0..63 (read by layer1 now AND by layer0 next step)
        a1[(r + hi8) * A1_ROW + jj * 16 + nl] = (_Float16)(ov * fast_tanh(cc));
      }
    }
    lds_order();

    // ---- layer 1 A fragments: A = [h0_t | h1_prev] ----
    v16h A1[4];
#pragma unroll
    for (int c = 0; c < 4; ++c)
      A1[c] = ld_frag2(a1 + nl * A1_ROW + c * 32 + 2 * hi8);

#pragma unroll
    for (int jj = 0; jj < 4; ++jj) {
      v8f acc[4];
      {
        v16h bA = ld_frag2(w1f + (size_t)((jj * 4 + 0) * 32 + lane) * 16);
        v16h bB = ld_frag2(w1f + (size_t)((jj * 4 + 1) * 32 + lane) * 16);
#pragma unroll
        for (int i = 0; i < 16; ++i) {
          const int g = i / 4, c = i - g * 4;
          v16h bN = bA;  // dead placeholder past the end
          if (i + 2 < 16) {
            const int i2 = i + 2, g2 = i2 / 4, c2 = i2 - g2 * 4;
            bN = ld_frag2(w1f + (size_t)(((jj + g2 * 4) * 4 + c2) * 32 + lane) * 16);
          }
          acc[g] = __builtin_amdgcn_wmma_f32_16x16x32_f16(
              false, A1[c], false, bA, (short)0, (c == 0) ? zero : acc[g], false, false);
          bA = bB; bB = bN;
        }
      }
#pragma unroll
      for (int r = 0; r < 8; ++r) {
        float iv = fast_sigmoid(acc[0][r] + bias1[jj]);
        float fv = fast_sigmoid(acc[1][r] + bias1[jj + 4]);
        float gv = fast_tanh(acc[2][r] + bias1[jj + 8]);
        float ov = fast_sigmoid(acc[3][r] + bias1[jj + 12]);
        float cc = fv * c1[jj * 8 + r] + iv * gv;
        c1[jj * 8 + r] = cc;
        a1[(r + hi8) * A1_ROW + 64 + jj * 16 + nl] = (_Float16)(ov * fast_tanh(cc));
      }
    }
    lds_order();
  }

  // ---- final FC: out[b] = h1_final . Wfc + bfc (all 32 lanes, shfl reduce) ----
  {
    const int kb = (lane & 16) ? 32 : 0;
    float s = 0.0f;
#pragma unroll
    for (int k = 0; k < 32; ++k)
      s += (float)a1[nl * A1_ROW + 64 + kb + k] * Wfc[kb + k];
    s += __shfl_xor(s, 16, 32);
    if (lane < 16) out[rowbase + nl] = s + bfc[0];
  }
}

extern "C" void kernel_launch(void* const* d_in, const int* in_sizes, int n_in,
                              void* d_out, int out_size, void* d_ws, size_t ws_size,
                              hipStream_t stream) {
  const float* x    = (const float*)d_in[0];
  const float* Wih0 = (const float*)d_in[1];
  const float* Whh0 = (const float*)d_in[2];
  const float* bih0 = (const float*)d_in[3];
  const float* bhh0 = (const float*)d_in[4];
  const float* Wih1 = (const float*)d_in[5];
  const float* Whh1 = (const float*)d_in[6];
  const float* bih1 = (const float*)d_in[7];
  const float* bhh1 = (const float*)d_in[8];
  const float* Wfc  = (const float*)d_in[9];
  const float* bfc  = (const float*)d_in[10];
  float* out = (float*)d_out;

  // 256 blocks x 1 wave32; each wave owns 16 batch rows and a full WGP's LDS bandwidth.
  hipLaunchKernelGGL(lstm_fused_wmma, dim3(B_TOT / 16), dim3(32), SMEM_BYTES, stream,
                     x, Wih0, Whh0, bih0, bhh0, Wih1, Whh1, bih1, bhh1, Wfc, bfc, out);
}
